// MixtralSparseMoeBlock_39711267618827
// MI455X (gfx1250) — compile-verified, matrix operand order checked by reading, not run on hardware
//
#include <hip/hip_runtime.h>
#include <math.h>

// ---------------------------------------------------------------------------
// Mixtral sparse MoE block for MI455X (gfx1250).
// Tokens = B*S = 4096, H = 1024, I = 4096, E = 8, TOP_K = 2.
//
// Roofline: dense expert math is 826 GFLOP; top-2 routing makes only
// ~206 GFLOP non-zero. Contiguous token tiles almost never skip (0.75^16),
// so tokens are GATHERED per expert (deterministic wave32 ballot scan) and
// each workgroup runs 16 same-expert tokens. Weights (402 MB fp32 ~ 17 us
// at 23.3 TB/s) are reused by ~64 tiles/expert, so fp32->bf16 conversion +
// WMMA-fragment swizzle is hoisted into one-time pre-pass kernels in d_ws;
// the hot loop feeds v_wmma_f32_16x16x32_bf16 from coalesced
// global_load_b128 fragment streams (2 loads/fragment, no converts).
// ---------------------------------------------------------------------------

#define TOKENS   4096
#define HDIM     1024
#define IDIM     4096
#define NEXP     8
#define TILE_M   16
#define XS_STRIDE 1032        // x tile stride (bf16), %4==0 for b64 stores
#define HS_STRIDE 130         // h tile stride (bf16)
#define FRAG_ELEMS 512        // one 16x16x32 bf16 fragment = 512 elems = 1KB

typedef __attribute__((ext_vector_type(16))) __bf16 v16bf;
typedef __attribute__((ext_vector_type(4)))  __bf16 v4bf;
typedef __attribute__((ext_vector_type(2)))  __bf16 v2bf;
typedef __attribute__((ext_vector_type(8)))  float  v8f;

// Per-lane K offset inside a 16x32 fragment (ISA 7.12.2 16-bit layout).
__device__ __host__ inline int frag_koff(int j, int l) {
  return ((j >> 2) << 4) + ((l >> 4) << 3) + ((j & 3) << 1);
}

// --- A fragment (16x32 bf16) from row-major LDS tile (k-pairs contiguous) --
__device__ inline v16bf load_a_lds(const __bf16* base, int stride, int row,
                                   int kbase, int k0) {
  v16bf a;
#pragma unroll
  for (int j = 0; j < 8; ++j) {
    int off = ((j >> 2) << 4) + kbase + ((j & 3) << 1);
    const __bf16* p = base + row * stride + k0 + off;
    a[2 * j]     = p[0];
    a[2 * j + 1] = p[1];
  }
  return a;
}

// ---------------------------------------------------------------------------
__global__ void zero_kernel(float* __restrict__ p, int n) {
  int i = blockIdx.x * 256 + threadIdx.x;
  if (i < n) p[i] = 0.0f;
}

// Router logits: logits[t][e] = dot(hidden[t,:], gate_w[:,e]) in fp32.
__global__ void router_kernel(const float* __restrict__ hidden,
                              const float* __restrict__ gate_w,
                              float* __restrict__ logits) {
  int tid   = threadIdx.x;                 // 32 tokens x 8 experts per block
  int token = blockIdx.x * 32 + (tid >> 3);
  int e     = tid & 7;
  const float* h = hidden + (size_t)token * HDIM;
  float acc = 0.0f;
  for (int k = 0; k < HDIM; ++k) acc = fmaf(h[k], gate_w[k * NEXP + e], acc);
  logits[(size_t)token * NEXP + e] = acc;
}

// Top-2 + softmax over the selected logits -> dense combine[t][e].
__global__ void top2_kernel(const float* __restrict__ logits,
                            float* __restrict__ combine) {
  int token = blockIdx.x * 256 + threadIdx.x;
  if (token >= TOKENS) return;
  float l[NEXP];
#pragma unroll
  for (int e = 0; e < NEXP; ++e) l[e] = logits[(size_t)token * NEXP + e];
  int i0 = 0;
#pragma unroll
  for (int e = 1; e < NEXP; ++e) if (l[e] > l[i0]) i0 = e;  // low idx wins ties
  int i1 = (i0 == 0) ? 1 : 0;
#pragma unroll
  for (int e = 0; e < NEXP; ++e)
    if (e != i0 && l[e] > l[i1]) i1 = e;
  float w = __expf(l[i1] - l[i0]);   // l[i0] >= l[i1]
  float s = 1.0f + w;
#pragma unroll
  for (int e = 0; e < NEXP; ++e) combine[(size_t)token * NEXP + e] = 0.0f;
  combine[(size_t)token * NEXP + i0] = 1.0f / s;
  combine[(size_t)token * NEXP + i1] = w / s;
}

// ---------------------------------------------------------------------------
// MoE dispatch: one wave32 per expert scans tokens IN ORDER (deterministic),
// compacting routed tokens + gate weights via ballot/popc prefix sums.
// Pads each list to a multiple of 16 with -1 sentinels; writes count[e].
__global__ void build_lists_kernel(const float* __restrict__ combine,
                                   int* __restrict__ gather,
                                   float* __restrict__ gweight,
                                   int* __restrict__ counts) {
  const int e    = blockIdx.x;
  const int lane = threadIdx.x;            // 32 threads (one wave)
  int base = 0;
  for (int t0 = 0; t0 < TOKENS; t0 += 32) {
    int tok = t0 + lane;
    float c = combine[(size_t)tok * NEXP + e];
    bool pred = c > 0.0f;
#if __has_builtin(__builtin_amdgcn_ballot_w32)
    unsigned mask = __builtin_amdgcn_ballot_w32(pred);
#else
    unsigned mask = (unsigned)__ballot(pred);
#endif
    int pos = base + __popc(mask & ((1u << lane) - 1u));
    if (pred) {
      gather[e * TOKENS + pos]  = tok;
      gweight[e * TOKENS + pos] = c;
    }
    base += __popc(mask);
  }
  int padded = (base + TILE_M - 1) & ~(TILE_M - 1);
  for (int i = base + lane; i < padded; i += 32) gather[e * TOKENS + i] = -1;
  if (lane == 0) counts[e] = base;
}

// ---------------------------------------------------------------------------
// One-time pre-pass: fp32 W[e][K][N] -> bf16 B-fragments in WMMA lane order.
// korder==0: f = (e*ntiles + n_tile)*ksteps + k_step   (GEMM1 streams over K)
// korder==1: f = (e*ksteps + k_step)*ntiles + n_tile   (GEMM2 streams over N)
__global__ void swizzle_b_kernel(const float* __restrict__ W,
                                 __bf16* __restrict__ out,
                                 int K, int N, int korder) {
  const int f  = blockIdx.x;
  const int pr = threadIdx.x;          // pair 0..255
  const int l  = pr >> 3;              // lane 0..31
  const int j  = pr & 7;               // vgpr pair 0..7
  const int ksteps = K / 32, ntiles = N / 16, fper = ksteps * ntiles;
  const int e = f / fper, r = f % fper;
  int n_tile, k_step;
  if (korder) { k_step = r / ntiles; n_tile = r % ntiles; }
  else        { n_tile = r / ksteps; k_step = r % ksteps; }
  const int off = frag_koff(j, l);
  const float* We = W + (size_t)e * K * N;
  size_t src = (size_t)(k_step * 32 + off) * N + n_tile * 16 + (l & 15);
  v2bf d = {(__bf16)We[src], (__bf16)We[src + N]};
  *(v2bf*)&out[(size_t)f * FRAG_ELEMS + l * 16 + 2 * j] = d;
}

// ---------------------------------------------------------------------------
// One workgroup = (16 gathered same-expert tokens, expert). 8 wave32s.
// out[toks] += gw * ( silu(x@w1[e]) * (x@w3[e]) ) @ w2[e]
// ---------------------------------------------------------------------------
__global__ __launch_bounds__(256)
void moe_expert_kernel(const float* __restrict__ hidden,
                       const __bf16* __restrict__ w1f,
                       const __bf16* __restrict__ w3f,
                       const __bf16* __restrict__ w2f,
                       const int* __restrict__ gather,
                       const float* __restrict__ gweight,
                       const int* __restrict__ counts,
                       float* __restrict__ out) {
  __shared__ __attribute__((aligned(16))) __bf16 xs[TILE_M * XS_STRIDE];
  __shared__ __attribute__((aligned(16))) __bf16 hs[TILE_M * HS_STRIDE];
  __shared__ int   toks[TILE_M];
  __shared__ float gws[TILE_M];

  const int tile = blockIdx.x;
  const int e    = blockIdx.y;
  if (tile * TILE_M >= counts[e]) return;   // past this expert's token list

  const int tid   = threadIdx.x;
  const int wave  = tid >> 5;
  const int lane  = tid & 31;
  const int nrow  = lane & 15;
  const int kbase = (lane >> 4) << 3;

  if (tid < TILE_M) {
    toks[tid] = gather[e * TOKENS + tile * TILE_M + tid];
    gws[tid]  = gweight[e * TOKENS + tile * TILE_M + tid];
  }
  __syncthreads();

  // Stage gathered 16x1024 fp32 rows -> bf16 LDS (b128 loads, b64 stores).
  {
    const int r = tid >> 4;                  // row 0..15
    const int c = (tid & 15) * 64;           // 64 floats per thread
    const int tok = toks[r];
    if (tok >= 0) {
      const float* srcr = hidden + (size_t)tok * HDIM + c;
#pragma unroll
      for (int v = 0; v < 16; ++v) {
        float4 f = ((const float4*)srcr)[v];
        v4bf p = {(__bf16)f.x, (__bf16)f.y, (__bf16)f.z, (__bf16)f.w};
        *(v4bf*)&xs[r * XS_STRIDE + c + 4 * v] = p;
      }
    } else {
      v4bf z = {(__bf16)0.0f, (__bf16)0.0f, (__bf16)0.0f, (__bf16)0.0f};
#pragma unroll
      for (int v = 0; v < 16; ++v) *(v4bf*)&xs[r * XS_STRIDE + c + 4 * v] = z;
    }
  }
  __syncthreads();

  v8f oacc[8];
#pragma unroll
  for (int t = 0; t < 8; ++t) oacc[t] = (v8f){};

#pragma unroll 1
  for (int chunk = 0; chunk < IDIM / 128; ++chunk) {
    // ---- GEMM1: this wave's 16x16 tile of silu(x@w1) * (x@w3) ------------
    const int ntile = chunk * 8 + wave;              // I-column tile
    const __bf16* f1 =
        w1f + (size_t)(e * (IDIM / 16) + ntile) * 32 * FRAG_ELEMS + lane * 16;
    const __bf16* f3 =
        w3f + (size_t)(e * (IDIM / 16) + ntile) * 32 * FRAG_ELEMS + lane * 16;
    if (chunk + 1 < IDIM / 128) {                    // global_prefetch_b8
      __builtin_prefetch(f1 + (size_t)(8 * 32) * FRAG_ELEMS, 0, 1);
      __builtin_prefetch(f3 + (size_t)(8 * 32) * FRAG_ELEMS, 0, 1);
    }
    v8f acc1 = (v8f){};
    v8f acc3 = (v8f){};
#pragma unroll 4
    for (int ks = 0; ks < 32; ++ks) {                // K = 1024 in 32 steps
      v16bf a  = load_a_lds(xs, XS_STRIDE, nrow, kbase, ks * 32);
      v16bf b1 = *(const v16bf*)(f1 + (size_t)ks * FRAG_ELEMS);
      v16bf b3 = *(const v16bf*)(f3 + (size_t)ks * FRAG_ELEMS);
      acc1 = __builtin_amdgcn_wmma_f32_16x16x32_bf16(false, a, false, b1,
                                                     (short)0, acc1, false, false);
      acc3 = __builtin_amdgcn_wmma_f32_16x16x32_bf16(false, a, false, b3,
                                                     (short)0, acc3, false, false);
    }
    // silu(g)*u -> bf16 h chunk. C-frag: M = j + 8*(lane>=16), N = nrow.
    const int lcol = wave * 16 + nrow;
#pragma unroll
    for (int j = 0; j < 8; ++j) {
      float g = acc1[j];
      float h = g * __builtin_amdgcn_rcpf(1.0f + __expf(-g)) * acc3[j];
      hs[(j + ((lane >> 4) << 3)) * HS_STRIDE + lcol] = (__bf16)h;
    }
    __syncthreads();

    // ---- GEMM2: out += h_chunk(16x128) @ w2[chunk rows](128xH) -----------
#pragma unroll 1
    for (int kk = 0; kk < 4; ++kk) {                 // 128 K in 4 steps
      v16bf a = load_a_lds(hs, HS_STRIDE, nrow, kbase, kk * 32);
      const __bf16* f2 =
          w2f + ((size_t)(e * (IDIM / 32) + chunk * 4 + kk) * (HDIM / 16) +
                 wave * 8) * FRAG_ELEMS + lane * 16;
#pragma unroll
      for (int t = 0; t < 8; ++t) {
        v16bf b = *(const v16bf*)(f2 + (size_t)t * FRAG_ELEMS);
        oacc[t] = __builtin_amdgcn_wmma_f32_16x16x32_bf16(false, a, false, b,
                                                          (short)0, oacc[t],
                                                          false, false);
      }
    }
    __syncthreads();   // before next chunk overwrites hs
  }

  // Scale rows by gate weight, scatter-accumulate with f32 atomics.
  int   ctok[8];
  float cscale[8];
#pragma unroll
  for (int j = 0; j < 8; ++j) {
    int M = j + ((lane >> 4) << 3);
    ctok[j]   = toks[M];
    cscale[j] = gws[M];
  }
#pragma unroll
  for (int t = 0; t < 8; ++t) {
    const int n0 = (wave * 8 + t) * 16;
#pragma unroll
    for (int j = 0; j < 8; ++j) {
      if (ctok[j] >= 0)
        atomicAdd(&out[(size_t)ctok[j] * HDIM + n0 + nrow],
                  oacc[t][j] * cscale[j]);
    }
  }
}

// ---------------------------------------------------------------------------
extern "C" void kernel_launch(void* const* d_in, const int* in_sizes, int n_in,
                              void* d_out, int out_size, void* d_ws,
                              size_t ws_size, hipStream_t stream) {
  (void)in_sizes; (void)n_in; (void)out_size; (void)ws_size;
  const float* hidden = (const float*)d_in[0];   // [B,S,H]
  const float* gate_w = (const float*)d_in[1];   // [H,E]
  const float* w1     = (const float*)d_in[2];   // [E,H,I]
  const float* w3     = (const float*)d_in[3];   // [E,H,I]
  const float* w2     = (const float*)d_in[4];   // [E,I,H]

  float* out_final = (float*)d_out;                         // TOKENS*H
  float* logits    = (float*)d_out + (size_t)TOKENS * HDIM; // TOKENS*E

  // Workspace layout (~192 MB fragments + dispatch metadata):
  const size_t WELEMS = (size_t)NEXP * HDIM * IDIM;         // 33,554,432
  __bf16* w1f = (__bf16*)d_ws;
  __bf16* w3f = w1f + WELEMS;
  __bf16* w2f = w3f + WELEMS;
  float* combine = (float*)(w2f + WELEMS);                  // TOKENS*E
  float* gweight = combine + (size_t)TOKENS * NEXP;         // E*TOKENS
  int*   gatherl = (int*)(gweight + (size_t)NEXP * TOKENS); // E*TOKENS
  int*   counts  = gatherl + (size_t)NEXP * TOKENS;         // E

  zero_kernel<<<(TOKENS * HDIM + 255) / 256, 256, 0, stream>>>(out_final,
                                                               TOKENS * HDIM);
  router_kernel<<<TOKENS / 32, 256, 0, stream>>>(hidden, gate_w, logits);
  top2_kernel<<<TOKENS / 256, 256, 0, stream>>>(logits, combine);
  build_lists_kernel<<<NEXP, 32, 0, stream>>>(combine, gatherl, gweight,
                                              counts);

  // One-time bf16 fragment swizzles (one block per 1KB fragment).
  const int NFRAG_W = NEXP * (IDIM / 16) * (HDIM / 32);     // 65536
  swizzle_b_kernel<<<NFRAG_W, 256, 0, stream>>>(w1, w1f, HDIM, IDIM, 0);
  swizzle_b_kernel<<<NFRAG_W, 256, 0, stream>>>(w3, w3f, HDIM, IDIM, 0);
  swizzle_b_kernel<<<NFRAG_W, 256, 0, stream>>>(w2, w2f, IDIM, HDIM, 1);

  dim3 grid(TOKENS / TILE_M, NEXP);   // worst-case; inactive tiles early-exit
  moe_expert_kernel<<<grid, 256, 0, stream>>>(hidden, w1f, w3f, w2f, gatherl,
                                              gweight, counts, out_final);
}
